// GraphNetBlock_33672543601340
// MI455X (gfx1250) — matrix operand order repaired and checked by reading, not compiled, not run on hardware
//
#include <hip/hip_runtime.h>

#define HDIM 128

typedef __attribute__((ext_vector_type(16))) __bf16        v16bf;
typedef __attribute__((ext_vector_type(8)))  float         v8f;
typedef __attribute__((ext_vector_type(16))) unsigned short v16u16;
typedef __attribute__((ext_vector_type(8)))  unsigned short v8u16;

static __device__ __forceinline__ unsigned short f2bf(float f) {
    unsigned int u = __float_as_uint(f);
    u += 0x7FFFu + ((u >> 16) & 1u);   // round-to-nearest-even
    return (unsigned short)(u >> 16);
}

// ---------------------------------------------------------------------------
// One 16-row strip x 128-col GEMM using v_wmma_f32_16x16x32_bf16.
// src: LDS activations, row-major bf16, stride sstride (elements).
// Wm : global bf16 weights pre-swizzled into B-fragment order:
//      frag (kb,nt) -> 32 lanes x 16 bf16; lane<16 holds col nt*16+lane K=kb*32+0..7,16..23,
//      lane>=16 holds same col K=kb*32+8..15,24..31.
// A-fragment per ISA 16-bit A 16x32 layout: lane L -> row m0+L%16,
//      K = base..base+7 and 16+base..16+base+7 with base = 8*(L>=16).
// ---------------------------------------------------------------------------
static __device__ __forceinline__ void gemm_rowstrip(
    v8f acc[8], const unsigned short* __restrict__ src, int sstride,
    const unsigned short* __restrict__ Wm, int kblocks, int m0, int lane)
{
    const int lhalf = lane >> 4, lmod = lane & 15;
    const unsigned short* rp = src + (m0 + lmod) * sstride + lhalf * 8;
    const v16u16* wp = (const v16u16*)Wm + lane;
    for (int kb = 0; kb < kblocks; ++kb) {
        v8u16 lo = *(const v8u16*)(rp + kb * 32);
        v8u16 hi = *(const v8u16*)(rp + kb * 32 + 16);
        v16u16 au;
#pragma unroll
        for (int i = 0; i < 8; ++i) { au[i] = lo[i]; au[i + 8] = hi[i]; }
        v16bf a = __builtin_bit_cast(v16bf, au);
#pragma unroll
        for (int t = 0; t < 8; ++t) {
            v16bf b = __builtin_bit_cast(v16bf, wp[(kb * 8 + t) * 32]);
            acc[t] = __builtin_amdgcn_wmma_f32_16x16x32_bf16(
                false, a, false, b, (short)0, acc[t], false, false);
        }
    }
}

static __device__ __forceinline__ void zero_acc(v8f acc[8]) {
#pragma unroll
    for (int t = 0; t < 8; ++t) { v8f z = {}; acc[t] = z; }
}

// C/D layout: VGPR r, lanes 0-15 -> M=r, lanes 16-31 -> M=8+r; N = t*16 + lane%16.
static __device__ __forceinline__ void bias_relu_store(
    const v8f acc[8], const float* __restrict__ bias,
    unsigned short* __restrict__ dst, int dstride, int m0, int lane)
{
    const int lhalf = lane >> 4, lmod = lane & 15;
#pragma unroll
    for (int t = 0; t < 8; ++t) {
        float bv = bias[t * 16 + lmod];
#pragma unroll
        for (int r = 0; r < 8; ++r) {
            float v = acc[t][r] + bv;
            v = v > 0.f ? v : 0.f;
            dst[(m0 + lhalf * 8 + r) * dstride + t * 16 + lmod] = f2bf(v);
        }
    }
}

// ---------------------------------------------------------------------------
// Edge MLP: gather [nf[snd] | nf[rcv] | ea] -> 4-layer MLP -> LN,
// write ea+upd to out_edge, scatter-add upd into agg via f32 atomics.
// ---------------------------------------------------------------------------
__global__ __launch_bounds__(256)
void edge_mlp(const float* __restrict__ nf, const float* __restrict__ ea,
              const int* __restrict__ ei, const unsigned short* __restrict__ W,
              const float* __restrict__ b0, const float* __restrict__ b1,
              const float* __restrict__ b2, const float* __restrict__ b3,
              const float* __restrict__ g,  const float* __restrict__ be,
              float* __restrict__ out_edge, float* __restrict__ agg, int E)
{
    constexpr int XS = 392, HS = 136;
    __shared__ __align__(16) unsigned short Xs[128 * XS];
    __shared__ __align__(16) unsigned short Hs[128 * HS];
    float* LNbuf = (float*)Xs;   // reused after all waves pass layer 0

    const int tid = threadIdx.x;
    const int e0  = blockIdx.x * 128;

    // ---- stage X tile: 3 segs x 128 rows x 32 float4 ----
    for (int it = 0; it < 48; ++it) {
        int idx = tid + it * 256;
        int seg = idx >> 12;
        int rem = idx & 4095;
        int row = rem >> 5;
        int c4  = rem & 31;
        int e   = e0 + row;
        uint2 pk; pk.x = 0u; pk.y = 0u;
        if (e < E) {
            const float* src;
            if (seg == 0)      src = nf + (size_t)ei[e] * HDIM;
            else if (seg == 1) src = nf + (size_t)ei[E + e] * HDIM;
            else               src = ea + (size_t)e * HDIM;
            float4 v = ((const float4*)src)[c4];
            pk.x = (unsigned)f2bf(v.x) | ((unsigned)f2bf(v.y) << 16);
            pk.y = (unsigned)f2bf(v.z) | ((unsigned)f2bf(v.w) << 16);
        }
        *(uint2*)(Xs + row * XS + seg * HDIM + c4 * 4) = pk;
    }
    __syncthreads();

    const int lane = tid & 31;
    const int m0   = (tid >> 5) * 16;
    const int lhalf = lane >> 4, lmod = lane & 15;

    v8f acc[8];
    zero_acc(acc);
    gemm_rowstrip(acc, Xs, XS, W, 12, m0, lane);                 // 384 -> 128
    bias_relu_store(acc, b0, Hs, HS, m0, lane);
    asm volatile("s_wait_dscnt 0x0" ::: "memory");

    zero_acc(acc);
    gemm_rowstrip(acc, Hs, HS, W + 49152, 4, m0, lane);          // 128 -> 128
    bias_relu_store(acc, b1, Hs, HS, m0, lane);
    asm volatile("s_wait_dscnt 0x0" ::: "memory");

    zero_acc(acc);
    gemm_rowstrip(acc, Hs, HS, W + 65536, 4, m0, lane);
    bias_relu_store(acc, b2, Hs, HS, m0, lane);
    asm volatile("s_wait_dscnt 0x0" ::: "memory");

    zero_acc(acc);
    gemm_rowstrip(acc, Hs, HS, W + 81920, 4, m0, lane);          // final, no relu

    __syncthreads();   // everyone past layer 0 -> safe to recycle Xs as LN buffer
#pragma unroll
    for (int t = 0; t < 8; ++t) {
        float bv = b3[t * 16 + lmod];
#pragma unroll
        for (int r = 0; r < 8; ++r)
            LNbuf[(m0 + lhalf * 8 + r) * 129 + t * 16 + lmod] = acc[t][r] + bv;
    }
    __syncthreads();

    if (tid < 128) {
        int e = e0 + tid;
        if (e < E) {
            const float* hrow = LNbuf + tid * 129;
            float mu = 0.f;
            for (int c = 0; c < HDIM; ++c) mu += hrow[c];
            mu *= (1.0f / HDIM);
            float var = 0.f;
            for (int c = 0; c < HDIM; ++c) { float d = hrow[c] - mu; var += d * d; }
            var *= (1.0f / HDIM);
            float rstd = rsqrtf(var + 1e-5f);
            int rcv = ei[E + e];
            float* aggr = agg + (size_t)rcv * HDIM;
            const float* earow = ea + (size_t)e * HDIM;
            float* orow = out_edge + (size_t)e * HDIM;
            for (int c = 0; c < HDIM; ++c) {
                float v = (hrow[c] - mu) * rstd * g[c] + be[c];
                orow[c] = earow[c] + v;
                unsafeAtomicAdd(aggr + c, v);   // global_atomic_add_f32
            }
        }
    }
}

// ---------------------------------------------------------------------------
// Node MLP: [nf | agg] -> 4-layer MLP -> LN, out = nf + upd.
// ---------------------------------------------------------------------------
__global__ __launch_bounds__(256)
void node_mlp(const float* __restrict__ nf, const float* __restrict__ agg,
              const unsigned short* __restrict__ W,
              const float* __restrict__ b0, const float* __restrict__ b1,
              const float* __restrict__ b2, const float* __restrict__ b3,
              const float* __restrict__ g,  const float* __restrict__ be,
              float* __restrict__ out_node, int N)
{
    constexpr int XS = 264, HS = 136;
    __shared__ __align__(16) unsigned short Xs[128 * XS];
    __shared__ __align__(16) unsigned short Hs[128 * HS];
    float* LNbuf = (float*)Xs;

    const int tid = threadIdx.x;
    const int n0  = blockIdx.x * 128;

    for (int it = 0; it < 32; ++it) {               // 2 segs x 128 rows x 32 float4
        int idx = tid + it * 256;
        int seg = idx >> 12;
        int rem = idx & 4095;
        int row = rem >> 5;
        int c4  = rem & 31;
        int n   = n0 + row;
        uint2 pk; pk.x = 0u; pk.y = 0u;
        if (n < N) {
            const float* src = (seg == 0) ? nf + (size_t)n * HDIM
                                          : agg + (size_t)n * HDIM;
            float4 v = ((const float4*)src)[c4];
            pk.x = (unsigned)f2bf(v.x) | ((unsigned)f2bf(v.y) << 16);
            pk.y = (unsigned)f2bf(v.z) | ((unsigned)f2bf(v.w) << 16);
        }
        *(uint2*)(Xs + row * XS + seg * HDIM + c4 * 4) = pk;
    }
    __syncthreads();

    const int lane = tid & 31;
    const int m0   = (tid >> 5) * 16;
    const int lhalf = lane >> 4, lmod = lane & 15;

    v8f acc[8];
    zero_acc(acc);
    gemm_rowstrip(acc, Xs, XS, W, 8, m0, lane);                  // 256 -> 128
    bias_relu_store(acc, b0, Hs, HS, m0, lane);
    asm volatile("s_wait_dscnt 0x0" ::: "memory");

    zero_acc(acc);
    gemm_rowstrip(acc, Hs, HS, W + 32768, 4, m0, lane);
    bias_relu_store(acc, b1, Hs, HS, m0, lane);
    asm volatile("s_wait_dscnt 0x0" ::: "memory");

    zero_acc(acc);
    gemm_rowstrip(acc, Hs, HS, W + 49152, 4, m0, lane);
    bias_relu_store(acc, b2, Hs, HS, m0, lane);
    asm volatile("s_wait_dscnt 0x0" ::: "memory");

    zero_acc(acc);
    gemm_rowstrip(acc, Hs, HS, W + 65536, 4, m0, lane);

    __syncthreads();
#pragma unroll
    for (int t = 0; t < 8; ++t) {
        float bv = b3[t * 16 + lmod];
#pragma unroll
        for (int r = 0; r < 8; ++r)
            LNbuf[(m0 + lhalf * 8 + r) * 129 + t * 16 + lmod] = acc[t][r] + bv;
    }
    __syncthreads();

    if (tid < 128) {
        int n = n0 + tid;
        if (n < N) {
            const float* hrow = LNbuf + tid * 129;
            float mu = 0.f;
            for (int c = 0; c < HDIM; ++c) mu += hrow[c];
            mu *= (1.0f / HDIM);
            float var = 0.f;
            for (int c = 0; c < HDIM; ++c) { float d = hrow[c] - mu; var += d * d; }
            var *= (1.0f / HDIM);
            float rstd = rsqrtf(var + 1e-5f);
            const float* nrow = nf + (size_t)n * HDIM;
            float* orow = out_node + (size_t)n * HDIM;
            for (int c = 0; c < HDIM; ++c) {
                float v = (hrow[c] - mu) * rstd * g[c] + be[c];
                orow[c] = nrow[c] + v;
            }
        }
    }
}

// f32 [K,128] weight -> bf16 B-fragment order. One block per (kb,nt) fragment.
__global__ __launch_bounds__(256)
void convert_weights(const float* __restrict__ src, unsigned short* __restrict__ dst)
{
    int frag = blockIdx.x;
    int kb = frag >> 3, nt = frag & 7;
    for (int i = threadIdx.x; i < 512; i += 256) {
        int lane = i >> 4, j = i & 15;
        int k = kb * 32 + (j >> 3) * 16 + (lane >> 4) * 8 + (j & 7);
        int n = nt * 16 + (lane & 15);
        dst[frag * 512 + i] = f2bf(src[k * HDIM + n]);
    }
}

__global__ __launch_bounds__(256)
void zero_f32(float* __restrict__ p, long n)
{
    long i = (long)blockIdx.x * blockDim.x + threadIdx.x;
    long stride = (long)gridDim.x * blockDim.x;
    for (; i < n; i += stride) p[i] = 0.f;
}

extern "C" void kernel_launch(void* const* d_in, const int* in_sizes, int n_in,
                              void* d_out, int out_size, void* d_ws, size_t ws_size,
                              hipStream_t stream)
{
    const float* nf = (const float*)d_in[0];
    const float* ea = (const float*)d_in[1];
    const int*   ei = (const int*)d_in[2];
    const int N = in_sizes[0] / HDIM;
    const int E = in_sizes[1] / HDIM;

    const float* e_w0 = (const float*)d_in[3];
    const float* e_b0 = (const float*)d_in[4];
    const float* e_w1 = (const float*)d_in[5];
    const float* e_b1 = (const float*)d_in[6];
    const float* e_w2 = (const float*)d_in[7];
    const float* e_b2 = (const float*)d_in[8];
    const float* e_w3 = (const float*)d_in[9];
    const float* e_b3 = (const float*)d_in[10];
    const float* e_g  = (const float*)d_in[11];
    const float* e_be = (const float*)d_in[12];
    const float* n_w0 = (const float*)d_in[13];
    const float* n_b0 = (const float*)d_in[14];
    const float* n_w1 = (const float*)d_in[15];
    const float* n_b1 = (const float*)d_in[16];
    const float* n_w2 = (const float*)d_in[17];
    const float* n_b2 = (const float*)d_in[18];
    const float* n_w3 = (const float*)d_in[19];
    const float* n_b3 = (const float*)d_in[20];
    const float* n_g  = (const float*)d_in[21];
    const float* n_be = (const float*)d_in[22];

    // workspace: swizzled bf16 weights then f32 agg buffer (16B aligned).
    unsigned short* wE = (unsigned short*)d_ws;            // 96+3*32 frags = 98304 ush
    unsigned short* wN = wE + 98304;                       // 64+3*32 frags = 81920 ush
    float* agg = (float*)((char*)d_ws + 360448);

    convert_weights<<<96, 256, 0, stream>>>(e_w0, wE);             // 384x128
    convert_weights<<<32, 256, 0, stream>>>(e_w1, wE + 49152);
    convert_weights<<<32, 256, 0, stream>>>(e_w2, wE + 65536);
    convert_weights<<<32, 256, 0, stream>>>(e_w3, wE + 81920);
    convert_weights<<<64, 256, 0, stream>>>(n_w0, wN);             // 256x128
    convert_weights<<<32, 256, 0, stream>>>(n_w1, wN + 32768);
    convert_weights<<<32, 256, 0, stream>>>(n_w2, wN + 49152);
    convert_weights<<<32, 256, 0, stream>>>(n_w3, wN + 65536);

    zero_f32<<<2048, 256, 0, stream>>>(agg, (long)N * HDIM);

    float* out_node = (float*)d_out;
    float* out_edge = (float*)d_out + (size_t)N * HDIM + (size_t)2 * E;

    edge_mlp<<<(E + 127) / 128, 256, 0, stream>>>(
        nf, ea, ei, wE, e_b0, e_b1, e_b2, e_b3, e_g, e_be, out_edge, agg, E);

    node_mlp<<<(N + 127) / 128, 256, 0, stream>>>(
        nf, agg, wN, n_b0, n_b1, n_b2, n_b3, n_g, n_be, out_node, N);

    // edge_index passthrough into the middle of the output tuple
    hipMemcpyAsync((float*)d_out + (size_t)N * HDIM, ei,
                   (size_t)2 * E * sizeof(int), hipMemcpyDeviceToDevice, stream);
}
